// WindowAttention_27822798143627
// MI455X (gfx1250) — matrix operand order repaired and checked by reading, not compiled, not run on hardware
//
#include <hip/hip_runtime.h>
#include <hip/hip_bf16.h>

typedef _Float16 v16h __attribute__((ext_vector_type(16)));
typedef _Float16 v8h  __attribute__((ext_vector_type(8)));
typedef _Float16 v4h  __attribute__((ext_vector_type(4)));
typedef float    v8f  __attribute__((ext_vector_type(8)));
typedef float    v4f  __attribute__((ext_vector_type(4)));
typedef unsigned int u32x4 __attribute__((ext_vector_type(4)));
typedef int          i32x4 __attribute__((ext_vector_type(4)));
typedef int          i32x8 __attribute__((ext_vector_type(8)));

union F16x16 { v16h v; v8h h[2]; _Float16 e[16]; };
union F16x4  { v4h  v; _Float16 e[4]; };
union F32x8  { v8f v; float e[8]; };

#define RS  264   // row stride (halfs): 528B = 33*16B, rotates LDS banks, 16B aligned
#define VTS 72    // row stride (halfs) for transposed V [256][64]

// ---------------- prep: weights -> f16, bias gather -> [H][64][64] f32 ----------------
__global__ __launch_bounds__(256) void prep_kernel(
    const float* __restrict__ qkv_w, const float* __restrict__ proj_w,
    const float* __restrict__ bias_table, const int* __restrict__ rel_index,
    _Float16* __restrict__ wq, _Float16* __restrict__ wp, float* __restrict__ biasT)
{
    int i = blockIdx.x * 256 + threadIdx.x;
    if (i < 196608) wq[i] = (_Float16)qkv_w[i];          // [768][256] row-major
    if (i < 65536)  wp[i] = (_Float16)proj_w[i];         // [256][256] row-major
    if (i < 32768) {                                     // biasT[h][m][n]
        int h  = i >> 12;
        int mn = i & 4095;                               // m*64+n
        biasT[i] = bias_table[rel_index[mn] * 8 + h];
    }
}

// ---- TDM: DMA one 32x256 f16 tile (global, row-major) into LDS with 264-half row stride
__device__ __forceinline__ void tdm_load_chunk(const _Float16* gsrc, unsigned lds_off)
{
    unsigned long long ga = (unsigned long long)(uintptr_t)gsrc;
    u32x4 g0;
    g0.x = 1u;                                            // count=1, user mode
    g0.y = lds_off;                                       // LDS dest byte offset
    g0.z = (unsigned)ga;                                  // global_addr[31:0]
    g0.w = (unsigned)((ga >> 32) & 0x1FFFFFFu) | (2u << 30);  // addr[56:32] | type=2
    i32x8 g1;
    g1[0] = (int)((1u << 16)    // data_size = 2 bytes
                | (1u << 20)    // pad_enable
                | (6u << 22)    // pad_interval: 128 dwords (one 512B row)
                | (3u << 25));  // pad_amount: 4 dwords (16B) -> 528B row stride
    g1[1] = (int)(256u << 16);  // tensor_dim0 = 256 (elements)
    g1[2] = (int)(32u  << 16);  // tensor_dim1 = 32
    g1[3] = (int)(256u << 16);  // tile_dim0 = 256
    g1[4] = 32;                 // tile_dim1 = 32
    g1[5] = 256;                // tensor_dim0_stride = 256 elements
    g1[6] = 0;
    g1[7] = 0;
    i32x4 z4 = {0, 0, 0, 0};
    i32x8 z8 = {0, 0, 0, 0, 0, 0, 0, 0};
    __builtin_amdgcn_tensor_load_to_lds(g0, g1, z4, z4, z8, 0);
}

// ---------------- fused window attention: 1 block == 1 window ----------------
__global__ __launch_bounds__(256) void win_attn_kernel(
    const float* __restrict__ x, const float* __restrict__ qkv_b,
    const float* __restrict__ proj_b, const _Float16* __restrict__ wq,
    const _Float16* __restrict__ wp, const float* __restrict__ biasT,
    float* __restrict__ out)
{
    __shared__ _Float16 sQ [64 * RS];     // Q (scaled) [token][h*32+d]
    __shared__ _Float16 sK [64 * RS];     // K          [token][h*32+d]
    __shared__ _Float16 sXO[64 * RS];     // x (ph1-2) then attn-out (ph3-4)
    __shared__ _Float16 sVT[256 * VTS];   // V transposed [h*32+d][token]
    __shared__ _Float16 sB [2][32 * RS];  // double-buffered staged weight chunk

    const int t    = threadIdx.x;
    const int lane = t & 31;
    const int wv   = t >> 5;              // 8 waves
    const int lc   = lane & 15;
    const int hb   = lane >> 4;
    const int blk  = blockIdx.x;

    // kick off first QKV weight chunk DMA immediately (overlaps x load)
    if (wv == 0) tdm_load_chunk(wq, (unsigned)(uintptr_t)&sB[0][0]);

    const float* xw = x + (size_t)blk * 64 * 256;

    // ---------- Phase 1: x -> LDS f16 (burst loads, then convert+store) ----------
    {
        v4f xr[16];
        #pragma unroll
        for (int it = 0; it < 16; ++it) {
            int idx4 = t + it * 256;                 // 4096 float4s
            xr[it] = *(const v4f*)(xw + ((idx4 >> 6) << 8) + ((idx4 & 63) << 2));
        }
        #pragma unroll
        for (int it = 0; it < 16; ++it) {
            int idx4 = t + it * 256;
            int r = idx4 >> 6, c4 = idx4 & 63;
            F16x4 p;
            p.e[0] = (_Float16)xr[it].x; p.e[1] = (_Float16)xr[it].y;
            p.e[2] = (_Float16)xr[it].z; p.e[3] = (_Float16)xr[it].w;
            *(v4h*)&sXO[r * RS + (c4 << 2)] = p.v;   // 8B aligned ds_store_b64
        }
    }
    __syncthreads();

    // per-wave GEMM role: M-tile = wv>>1, N-subtile = wv&1
    const int mt2 = wv >> 1, nw2 = wv & 1;

    // cache full A-strip of x (16 rows x 256 K) in 64 VGPRs
    F16x16 aX[8];
    {
        int r = mt2 * 16 + lc;
        #pragma unroll
        for (int kt = 0; kt < 8; ++kt) {
            aX[kt].h[0] = *(const v8h*)&sXO[r * RS + kt * 32 + hb * 8];
            aX[kt].h[1] = *(const v8h*)&sXO[r * RS + kt * 32 + 16 + hb * 8];
        }
    }

    const float qscale = 0.1767766952966369f;        // 1/sqrt(32)

    // ---------- Phase 2: QKV GEMM (64x256 @ 256x768), 24 chunks of 32 cols ----------
    for (int nc = 0; nc < 24; ++nc) {
        __syncthreads();                             // everyone done with buf^1
        if (wv == 0) {
            if (nc + 1 < 24) {
                tdm_load_chunk(wq + (size_t)(nc + 1) * 8192,
                               (unsigned)(uintptr_t)&sB[(nc + 1) & 1][0]);
                __builtin_amdgcn_s_wait_tensorcnt(1);   // chunk nc arrived
            } else {
                __builtin_amdgcn_s_wait_tensorcnt(0);
            }
        }
        __syncthreads();                             // chunk nc visible to all
        const _Float16* sb = sB[nc & 1];

        F16x16 bf[8];
        #pragma unroll
        for (int kt = 0; kt < 8; ++kt) {             // burst-load all B fragments
            bf[kt].h[0] = *(const v8h*)&sb[(nw2 * 16 + lc) * RS + kt * 32 + hb * 8];
            bf[kt].h[1] = *(const v8h*)&sb[(nw2 * 16 + lc) * RS + kt * 32 + 16 + hb * 8];
        }
        F32x8 acc; acc.v = (v8f){0,0,0,0,0,0,0,0};
        #pragma unroll
        for (int kt = 0; kt < 8; ++kt)
            acc.v = __builtin_amdgcn_wmma_f32_16x16x32_f16(
                false, aX[kt].v, false, bf[kt].v, (short)0, acc.v, false, false);

        int colbase = nc * 32 + nw2 * 16;            // wave-uniform
        int col = colbase + lc;
        float bb = qkv_b[col];
        int seg = colbase >> 8;                      // 0:Q 1:K 2:V (uniform branch)
        if (seg == 0) {
            #pragma unroll
            for (int v = 0; v < 8; ++v) {
                int m = mt2 * 16 + v + hb * 8;
                sQ[m * RS + col] = (_Float16)((acc.e[v] + bb) * qscale);
            }
        } else if (seg == 1) {
            #pragma unroll
            for (int v = 0; v < 8; ++v) {
                int m = mt2 * 16 + v + hb * 8;
                sK[m * RS + (col - 256)] = (_Float16)(acc.e[v] + bb);
            }
        } else {
            #pragma unroll
            for (int v = 0; v < 8; ++v) {
                int m = mt2 * 16 + v + hb * 8;
                sVT[(col - 512) * VTS + m] = (_Float16)(acc.e[v] + bb);  // V transposed
            }
        }
    }
    __syncthreads();

    // prefetch first proj chunk via TDM; it flies during the whole attention phase
    if (wv == 0) tdm_load_chunk(wp, (unsigned)(uintptr_t)&sB[0][0]);

    // ---------- Phase 3: attention, one wave per head ----------
    const int h = wv;

    F16x16 kf[4], qf[4];
    #pragma unroll
    for (int nt = 0; nt < 4; ++nt) {                 // A = K rows (tokens)
        int r = nt * 16 + lc;
        kf[nt].h[0] = *(const v8h*)&sK[r * RS + h * 32 + hb * 8];
        kf[nt].h[1] = *(const v8h*)&sK[r * RS + h * 32 + 16 + hb * 8];
    }
    #pragma unroll
    for (int mt = 0; mt < 4; ++mt) {                 // B = Q^T columns (tokens)
        int m = mt * 16 + lc;
        qf[mt].h[0] = *(const v8h*)&sQ[m * RS + h * 32 + hb * 8];
        qf[mt].h[1] = *(const v8h*)&sQ[m * RS + h * 32 + 16 + hb * 8];
    }

    F32x8 S[4][4];                                   // S^T tiles: rows=n(keys), cols=m(queries)
    #pragma unroll
    for (int nt = 0; nt < 4; ++nt)
        #pragma unroll
        for (int mt = 0; mt < 4; ++mt) {
            v8f z = {0,0,0,0,0,0,0,0};
            S[nt][mt].v = __builtin_amdgcn_wmma_f32_16x16x32_f16(
                false, kf[nt].v, false, qf[mt].v, (short)0, z, false, false);
        }

    // add relative-position bias: S^T[n][m] += biasT[h][m][n]
    #pragma unroll
    for (int mt = 0; mt < 4; ++mt) {
        int m = mt * 16 + lc;
        const float* bp = biasT + ((h * 64 + m) << 6);
        #pragma unroll
        for (int nt = 0; nt < 4; ++nt) {
            v4f b0 = *(const v4f*)(bp + nt * 16 + hb * 8);
            v4f b1 = *(const v4f*)(bp + nt * 16 + hb * 8 + 4);
            float bb[8] = {b0.x, b0.y, b0.z, b0.w, b1.x, b1.y, b1.z, b1.w};
            #pragma unroll
            for (int v = 0; v < 8; ++v) S[nt][mt].e[v] += bb[v];
        }
    }

    // softmax over keys: lane-local 32 values + partner half via shfl_xor(16)
    #pragma unroll
    for (int mt = 0; mt < 4; ++mt) {
        float mx = -3.0e38f;
        #pragma unroll
        for (int nt = 0; nt < 4; ++nt)
            #pragma unroll
            for (int v = 0; v < 8; ++v) mx = fmaxf(mx, S[nt][mt].e[v]);
        mx = fmaxf(mx, __shfl_xor(mx, 16, 32));
        float sum = 0.f;
        #pragma unroll
        for (int nt = 0; nt < 4; ++nt)
            #pragma unroll
            for (int v = 0; v < 8; ++v) {
                float p = __expf(S[nt][mt].e[v] - mx);
                S[nt][mt].e[v] = p;
                sum += p;
            }
        sum += __shfl_xor(sum, 16, 32);
        float inv = 1.0f / sum;
        #pragma unroll
        for (int nt = 0; nt < 4; ++nt)
            #pragma unroll
            for (int v = 0; v < 8; ++v) S[nt][mt].e[v] *= inv;
    }

    // O = P @ V ; P A-fragments come straight out of S^T registers (no transpose)
    F16x16 vf[2][2];
    #pragma unroll
    for (int kt = 0; kt < 2; ++kt)
        #pragma unroll
        for (int dt = 0; dt < 2; ++dt) {
            int vr = h * 32 + dt * 16 + lc;
            vf[kt][dt].h[0] = *(const v8h*)&sVT[vr * VTS + kt * 32 + hb * 8];
            vf[kt][dt].h[1] = *(const v8h*)&sVT[vr * VTS + kt * 32 + 16 + hb * 8];
        }
    F32x8 O[4][2];
    #pragma unroll
    for (int mt = 0; mt < 4; ++mt)
        #pragma unroll
        for (int dt = 0; dt < 2; ++dt) O[mt][dt].v = (v8f){0,0,0,0,0,0,0,0};

    #pragma unroll
    for (int mt = 0; mt < 4; ++mt) {
        #pragma unroll
        for (int kt = 0; kt < 2; ++kt) {
            F16x16 p;
            #pragma unroll
            for (int e = 0; e < 8; ++e) {
                p.e[e]     = (_Float16)S[2 * kt][mt].e[e];
                p.e[e + 8] = (_Float16)S[2 * kt + 1][mt].e[e];
            }
            #pragma unroll
            for (int dt = 0; dt < 2; ++dt) {
                O[mt][dt].v = __builtin_amdgcn_wmma_f32_16x16x32_f16(
                    false, p.v, false, vf[kt][dt].v, (short)0, O[mt][dt].v, false, false);
            }
        }
    }

    // attention output -> sXO (x buffer is dead)
    #pragma unroll
    for (int mt = 0; mt < 4; ++mt)
        #pragma unroll
        for (int dt = 0; dt < 2; ++dt) {
            int col = h * 32 + dt * 16 + lc;
            #pragma unroll
            for (int v = 0; v < 8; ++v) {
                int m = mt * 16 + v + hb * 8;
                sXO[m * RS + col] = (_Float16)O[mt][dt].e[v];
            }
        }
    __syncthreads();

    // ---------- Phase 4: output projection (64x256 @ 256x256) + bias ----------
    F16x16 aO[8];
    {
        int r = mt2 * 16 + lc;
        #pragma unroll
        for (int kt = 0; kt < 8; ++kt) {
            aO[kt].h[0] = *(const v8h*)&sXO[r * RS + kt * 32 + hb * 8];
            aO[kt].h[1] = *(const v8h*)&sXO[r * RS + kt * 32 + 16 + hb * 8];
        }
    }
    float* op = out + (size_t)blk * 64 * 256;
    for (int nc = 0; nc < 8; ++nc) {
        __syncthreads();
        if (wv == 0) {
            if (nc + 1 < 8) {
                tdm_load_chunk(wp + (size_t)(nc + 1) * 8192,
                               (unsigned)(uintptr_t)&sB[(nc + 1) & 1][0]);
                __builtin_amdgcn_s_wait_tensorcnt(1);
            } else {
                __builtin_amdgcn_s_wait_tensorcnt(0);
            }
        }
        __syncthreads();
        const _Float16* sb = sB[nc & 1];

        F16x16 bf[8];
        #pragma unroll
        for (int kt = 0; kt < 8; ++kt) {
            bf[kt].h[0] = *(const v8h*)&sb[(nw2 * 16 + lc) * RS + kt * 32 + hb * 8];
            bf[kt].h[1] = *(const v8h*)&sb[(nw2 * 16 + lc) * RS + kt * 32 + 16 + hb * 8];
        }
        F32x8 acc; acc.v = (v8f){0,0,0,0,0,0,0,0};
        #pragma unroll
        for (int kt = 0; kt < 8; ++kt)
            acc.v = __builtin_amdgcn_wmma_f32_16x16x32_f16(
                false, aO[kt].v, false, bf[kt].v, (short)0, acc.v, false, false);

        int col = nc * 32 + nw2 * 16 + lc;
        float pb = proj_b[col];
        #pragma unroll
        for (int v = 0; v < 8; ++v) {
            int m = mt2 * 16 + v + hb * 8;
            op[m * 256 + col] = acc.e[v] + pb;
        }
    }
}

// ---------------- host launcher ----------------
extern "C" void kernel_launch(void* const* d_in, const int* in_sizes, int n_in,
                              void* d_out, int out_size, void* d_ws, size_t ws_size,
                              hipStream_t stream)
{
    const float* x          = (const float*)d_in[0];
    const float* qkv_w      = (const float*)d_in[1];
    const float* qkv_b      = (const float*)d_in[2];
    const float* proj_w     = (const float*)d_in[3];
    const float* proj_b     = (const float*)d_in[4];
    const float* bias_table = (const float*)d_in[5];
    const int*   rel_index  = (const int*)d_in[6];
    float* out = (float*)d_out;

    _Float16* wq16  = (_Float16*)d_ws;              // 196608 halfs
    _Float16* wp16  = wq16 + 196608;                // 65536 halfs
    float*    biasT = (float*)(wp16 + 65536);       // 32768 floats (512KB offset, aligned)

    prep_kernel<<<768, 256, 0, stream>>>(qkv_w, proj_w, bias_table, rel_index,
                                         wq16, wp16, biasT);
    win_attn_kernel<<<4096, 256, 0, stream>>>(x, qkv_b, proj_b, wq16, wp16,
                                              biasT, out);
    (void)in_sizes; (void)n_in; (void)out_size; (void)ws_size;
}
